// KAN_Convolutional_Layer_1D_76922864271828
// MI455X (gfx1250) — matrix operand order repaired and checked by reading, not compiled, not run on hardware
//
#include <hip/hip_runtime.h>

// ---------------------------------------------------------------------------
// KAN 1-D conv, CDNA5 (gfx1250) WMMA implementation, round 2.
//
//   B=32, I=8, W=8192, O=32, K=5  ->  Wo = 8188
//   features per sample: 1 SiLU + 8 cubic B-spline bases = 9 (padded to 12)
//   reduction per tap: 8*12 = 96 rows ; 5 taps -> 15 K32-WMMA per 16x16 tile
//
// Round-2 changes:
//   * closed-form uniform cubic B-spline (4 non-zero bases, LDS scatter)
//     instead of full Cox-de-Boor recursion  (~7x less VALU in hot stage)
//   * WT=128: halves per-WG weight-copy + halo overhead; each wave owns two
//     16x16 output tiles sharing A fragments -> two independent WMMA chains
// ---------------------------------------------------------------------------

typedef _Float16 v16h __attribute__((ext_vector_type(16)));
typedef _Float16 v8h  __attribute__((ext_vector_type(8)));
typedef float    v8f  __attribute__((ext_vector_type(8)));

constexpr int BATCH = 32;
constexpr int ICH   = 8;
constexpr int WIDTH = 8192;
constexpr int OCH   = 32;
constexpr int KTAP  = 5;
constexpr int WOUT  = WIDTH - (KTAP - 1);   // 8188
constexpr int WT    = 128;                  // output positions per workgroup
constexpr int PPAD  = WT + KTAP - 1;        // 132 positions incl. halo
constexpr int ROWSP = 104;                  // padded reduction rows per tap (96 real)
constexpr int NW_H  = OCH * KTAP * ROWSP;   // combined-weight halves = 16640

// Fuse base_weight and spline_weight*spline_scaler into one padded f16
// tensor: wc[o][k][row], row = i*12 + c (c=0 SiLU, 1..8 spline, 9..11 zero),
// rows 96..103 zero padding.  Stride 104 halves -> 16B-aligned rows and
// conflict-free LDS access in the main kernel.
__global__ void kan_prep_weights(const float* __restrict__ bw,
                                 const float* __restrict__ sw,
                                 const float* __restrict__ ss,
                                 _Float16* __restrict__ wc) {
    int idx = blockIdx.x * 256 + threadIdx.x;
    if (idx >= NW_H) return;
    int o   = idx / (KTAP * ROWSP);
    int rem = idx % (KTAP * ROWSP);
    int k   = rem / ROWSP;
    int r   = rem % ROWSP;
    float v = 0.0f;
    if (r < 96) {
        int i = r / 12, c = r % 12;
        int w3 = (o * ICH + i) * KTAP + k;           // [o][i][k] flat
        if (c == 0)       v = bw[w3];
        else if (c <= 8)  v = sw[w3 * 8 + (c - 1)] * ss[w3];
    }
    wc[idx] = (_Float16)v;
}

__global__ __launch_bounds__(256)
void kan_conv_wmma(const float* __restrict__ x,
                   const _Float16* __restrict__ wc,
                   float* __restrict__ out) {
    __shared__ __align__(16) _Float16 sW[NW_H];          // 33280 B
    __shared__ __align__(16) _Float16 sF[PPAD * ROWSP];  // 27456 B

    const int tid = threadIdx.x;
    const int b   = blockIdx.y;
    const int w0  = blockIdx.x * WT;

    // ---- stage 1: combined weights (L2-hot) -> LDS, 16B vector copies ----
    {
        const uint4* src = (const uint4*)wc;
        uint4*       dst = (uint4*)sW;
        for (int i = tid; i < NW_H / 8; i += 256) dst[i] = src[i];
    }

    // ---- stage 2: featurize halo tile into LDS (f16) ----
    // Uniform knots t[j] = 0.4*j - 2.2: cubic B-spline has exactly 4
    // non-zero bases; closed-form blending + LDS scatter.
    for (int item = tid; item < ICH * PPAD; item += 256) {
        int i = item / PPAD;               // input channel (coalesced in p)
        int p = item % PPAD;               // local position
        int g = w0 + p;
        float xv = (g < WIDTH) ? x[((b * ICH + i) * WIDTH) + g] : 0.0f;

        _Float16* row = &sF[p * ROWSP + i * 12];
#pragma unroll
        for (int c = 0; c < 12; ++c) row[c] = (_Float16)0.0f;

        // SiLU channel
        float sig = 1.0f / (1.0f + __expf(-xv));
        row[0] = (_Float16)(xv * sig);

        // cell index and local coordinate in knot units (h = 0.4)
        float uu = (xv + 2.2f) * 2.5f;
        float fm = floorf(uu);
        int   m  = (int)fm;
        float u  = uu - fm;
        float um = 1.0f - u;
        float Bv[4];
        Bv[0] = um * um * um * (1.0f / 6.0f);
        Bv[1] = ((3.0f * u - 6.0f) * u * u + 4.0f) * (1.0f / 6.0f);
        Bv[2] = (((-3.0f * u + 3.0f) * u + 3.0f) * u + 1.0f) * (1.0f / 6.0f);
        Bv[3] = u * u * u * (1.0f / 6.0f);
#pragma unroll
        for (int t = 0; t < 4; ++t) {
            int j = m - 3 + t;             // degree-3 basis index
            if ((unsigned)j < 8u) row[1 + j] = (_Float16)Bv[t];
        }
    }
    // zero the pad rows 96..103 of every position (0 * garbage = NaN hazard)
    for (int p = tid; p < PPAD; p += 256) {
        _Float16* row = &sF[p * ROWSP + 96];
#pragma unroll
        for (int c = 0; c < 8; ++c) row[c] = (_Float16)0.0f;
    }
    __syncthreads();

    // ---- stage 3: WMMA.  8 waves; wave owns two 16x16 tiles sharing A ----
    const int lane  = tid & 31;
    const int wave  = tid >> 5;
    const int mt    = wave & 1;          // output-channel tile (o base = mt*16)
    const int nt    = wave >> 1;         // 0..3
    const int n0    = nt * 16;           // first position tile
    const int n1    = (nt + 4) * 16;     // second position tile
    const int lhalf = lane >> 4;         // 0: lanes 0-15, 1: lanes 16-31
    const int l15   = lane & 15;

    v8f acc0 = {0.f, 0.f, 0.f, 0.f, 0.f, 0.f, 0.f, 0.f};
    v8f acc1 = {0.f, 0.f, 0.f, 0.f, 0.f, 0.f, 0.f, 0.f};
    const int oRow = mt * 16 + l15;      // A-matrix row M = lane&15

#pragma unroll
    for (int k = 0; k < KTAP; ++k) {
        // A (16x32 f16): lane holds row M; lanes 0-15 K=0..15, lanes 16-31
        // K=16..31 -> two contiguous 16B runs.
        const _Float16* abase  = &sW[(oRow * KTAP + k) * ROWSP + lhalf * 16];
        // B (32x16 f16): lanes 0-15 hold col N with K=0..15, lanes 16-31
        // K=16..31 -> 32 contiguous bytes per lane in [pos][row] layout.
        const _Float16* bbase0 = &sF[(n0 + l15 + k) * ROWSP + lhalf * 16];
        const _Float16* bbase1 = &sF[(n1 + l15 + k) * ROWSP + lhalf * 16];
#pragma unroll
        for (int j = 0; j < 3; ++j) {        // 3 chunks of 32 reduction rows
            v8h a0  = *(const v8h*)(abase  + j * 32);
            v8h a1  = *(const v8h*)(abase  + j * 32 + 8);
            v8h b00 = *(const v8h*)(bbase0 + j * 32);
            v8h b01 = *(const v8h*)(bbase0 + j * 32 + 8);
            v8h b10 = *(const v8h*)(bbase1 + j * 32);
            v8h b11 = *(const v8h*)(bbase1 + j * 32 + 8);
            v16h A, B0m, B1m;
#pragma unroll
            for (int t = 0; t < 8; ++t) {
                A[t]   = a0[t];  A[t + 8]   = a1[t];
                B0m[t] = b00[t]; B0m[t + 8] = b01[t];
                B1m[t] = b10[t]; B1m[t + 8] = b11[t];
            }
            acc0 = __builtin_amdgcn_wmma_f32_16x16x32_f16(
                false, A, false, B0m, (short)0, acc0, false, false);
            acc1 = __builtin_amdgcn_wmma_f32_16x16x32_f16(
                false, A, false, B1m, (short)0, acc1, false, false);
        }
    }

    // ---- stage 4: store per C/D layout: VGPR v -> M = v + 8*lhalf, N = l15
    const int obase = mt * 16 + lhalf * 8;
    {
        const int w = w0 + n0 + l15;
        if (w < WOUT) {
#pragma unroll
            for (int v = 0; v < 8; ++v)
                out[((b * OCH + obase + v) * WOUT) + w] = acc0[v];
        }
    }
    {
        const int w = w0 + n1 + l15;
        if (w < WOUT) {
#pragma unroll
            for (int v = 0; v < 8; ++v)
                out[((b * OCH + obase + v) * WOUT) + w] = acc1[v];
        }
    }
}

extern "C" void kernel_launch(void* const* d_in, const int* in_sizes, int n_in,
                              void* d_out, int out_size, void* d_ws, size_t ws_size,
                              hipStream_t stream) {
    const float* x  = (const float*)d_in[0];  // (32, 8, 8192)
    const float* bw = (const float*)d_in[1];  // (32, 8, 5)
    const float* sw = (const float*)d_in[2];  // (32, 8, 5, 8)
    const float* ss = (const float*)d_in[3];  // (32, 8, 5)
    float* out = (float*)d_out;               // (32, 32, 8188)
    _Float16* wc = (_Float16*)d_ws;           // 33280 B combined weights

    kan_prep_weights<<<(NW_H + 255) / 256, 256, 0, stream>>>(bw, sw, ss, wc);

    dim3 grid(WIDTH / WT, BATCH);             // 64 x 32 workgroups
    kan_conv_wmma<<<grid, 256, 0, stream>>>(x, wc, out);
}
